// InvertedResidual_70222715289956
// MI455X (gfx1250) — matrix operand order repaired and checked by reading, not compiled, not run on hardware
//
#include <hip/hip_runtime.h>
#include <hip/hip_bf16.h>
#include <math.h>

// ---------- CDNA5 types ----------
typedef __attribute__((ext_vector_type(16))) __bf16        v16bf;
typedef __attribute__((ext_vector_type(8)))  float         v8f;
typedef __attribute__((ext_vector_type(4)))  unsigned int  u32x4;

#define EPSV 1e-5f

// problem dims
constexpr int Bn  = 64;
constexpr int Cc  = 64;
constexpr int Hh  = 56;
constexpr int HW  = 56 * 56;   // 3136
constexpr int HID = 384;
constexpr int OUP = 64;

// ---------- LDS layout (bytes), all fragment rows 16B-aligned & padded ----------
constexpr int HSTR  = 784;                    // h row stride: 384 bf16 + 8 pad = 392 ushorts
constexpr int OFF_H  = 0;                     // h: 256 pixels x HSTR = 200704
constexpr int W1STR = 144;                    // w1 row: 64 bf16 + 8 pad
constexpr int OFF_W1 = 256 * HSTR;            // 200704
constexpr int ASTR  = 144;                    // x-tile row: 64 bf16 + 8 pad
constexpr int KTSTR = 784;                    // kT row: 384 bf16 + 8 pad (== padded ws layout)
constexpr int OFF_U  = OFF_W1 + HID * W1STR;  // 256000 ; union{ A-tile | wd+bn2 | kT }
constexpr int SMEM_BYTES = OFF_U + OUP * KTSTR; // 306176  (< 320KB WGP LDS)

// ws kT: per sample 64 rows x 392 ushorts (padded rows, b128-copyable flat)
constexpr int KT_ROW_US   = KTSTR / 2;        // 392
constexpr int KT_PER_B_US = OUP * KT_ROW_US;  // 25088 ushorts = 50176 bytes
constexpr int KT_PER_B_BY = OUP * KTSTR;      // 50176

// ---------- helpers ----------
__device__ __forceinline__ unsigned short f2bf(float f) {
    __bf16 h = (__bf16)f;                     // native v_cvt (RNE) on gfx1250
    return __builtin_bit_cast(unsigned short, h);
}
__device__ __forceinline__ float bf2f(unsigned short h) {
    return __uint_as_float(((unsigned)h) << 16);
}

union FragU { u32x4 q[2]; v16bf v; };

__device__ __forceinline__ v16bf lds_frag2(const char* p0, const char* p1) {
    FragU u;
    u.q[0] = *(const u32x4*)p0;               // ds_load_b128
    u.q[1] = *(const u32x4*)p1;               // ds_load_b128
    return u.v;
}

__device__ __forceinline__ v8f wmma_bf16(v16bf a, v16bf b, v8f c) {
    return __builtin_amdgcn_wmma_f32_16x16x32_bf16(
        /*neg_a=*/false, a, /*neg_b=*/false, b,
        /*c_mod=*/(short)0, c, /*reuse_a=*/false, /*reuse_b=*/false);
}

// CDNA5 async copy: global -> LDS, tracked by ASYNCcnt (no VGPR data path).
// ldsa = LDS byte address (VDST), goff = 32-bit per-lane offset, base = 64-bit SGPR base.
__device__ __forceinline__ void async_g2l_b128(unsigned ldsa, unsigned goff,
                                               unsigned long long base) {
    asm volatile("global_load_async_to_lds_b128 %0, %1, %2"
                 :: "v"(ldsa), "v"(goff), "s"(base) : "memory");
}
__device__ __forceinline__ void wait_asynccnt0() {
    asm volatile("s_wait_asynccnt 0x0" ::: "memory");
}

// =====================================================================
// Kernel 1: router — pooled mean -> sigmoid gates -> per-sample kT (bf16)
// kT layout in ws: [b][o][392]  (padded rows == LDS layout, flat-copyable)
// =====================================================================
__global__ __launch_bounds__(256) void router_kernel(
    const float* __restrict__ x, const float* __restrict__ wr,
    const float* __restrict__ br, const float* __restrict__ we,
    unsigned short* __restrict__ kt)
{
    __shared__ float red[256];
    __shared__ float ps[64];
    __shared__ float rg[8];

    const int b = blockIdx.x;
    const int t = threadIdx.x;
    const int c = t & 63, seg = t >> 6;

    const float* xp = x + (size_t)(b * Cc + c) * HW + seg * 784;
    float s = 0.f;
    for (int i = 0; i < 784; ++i) s += xp[i];
    red[t] = s;
    __syncthreads();
    if (t < 64) ps[t] = (red[t] + red[t + 64] + red[t + 128] + red[t + 192]) * (1.f / 3136.f);
    __syncthreads();
    if (t < 8) {
        float z = br[t];
        for (int cc = 0; cc < 64; ++cc) z += ps[cc] * wr[t * 64 + cc];
        rg[t] = 1.f / (1.f + __expf(-z));
    }
    __syncthreads();
    for (int i = t; i < OUP * HID; i += 256) {
        const int o = i / HID, ch = i - o * HID;
        float a = 0.f;
#pragma unroll
        for (int e = 0; e < 8; ++e) a += rg[e] * we[e * OUP * HID + i];
        kt[(size_t)b * KT_PER_B_US + o * KT_ROW_US + ch] = f2bf(a);
    }
}

// =====================================================================
// Kernel 2: fused expand(WMMA) + BN1/ReLU6 + depthwise3x3 + BN2/ReLU6
//           + CondConv(WMMA) + BN3 + residual.
// grid = B * 16 tiles (4x4 tiles of 14x14 outputs, 16x16 input halo).
// =====================================================================
__global__ __launch_bounds__(256) void fused_kernel(
    const float* __restrict__ x,  const float* __restrict__ w1,
    const float* __restrict__ g1, const float* __restrict__ b1,
    const float* __restrict__ m1, const float* __restrict__ v1,
    const float* __restrict__ wd,
    const float* __restrict__ g2, const float* __restrict__ b2,
    const float* __restrict__ m2, const float* __restrict__ v2,
    const unsigned short* __restrict__ kt,
    const float* __restrict__ g3, const float* __restrict__ b3,
    const float* __restrict__ m3, const float* __restrict__ v3,
    float* __restrict__ out)
{
    extern __shared__ char sm[];
    unsigned short* hs = (unsigned short*)sm;          // h tile, stride 392 ushorts

    const int b   = blockIdx.x >> 4;
    const int tl  = blockIdx.x & 15;
    const int oy0 = (tl >> 2) * 14;
    const int ox0 = (tl & 3)  * 14;

    const int tid  = threadIdx.x;
    const int wv   = tid >> 5;
    const int lane = tid & 31;
    const int lrow = lane & 15;   // fragment row/col index
    const int kh   = lane >> 4;   // which K half this lane holds

    __builtin_prefetch(w1, 0, 1);                             // global_prefetch_b8
    __builtin_prefetch(kt + (size_t)b * KT_PER_B_US, 0, 1);

    // ---------------- load x tile (zero-padded) + w1 into LDS as bf16 -------------
    {
        const int p  = tid;               // pixel 0..255 in 16x16 halo tile
        const int py = p >> 4, px = p & 15;
        const int Gy = oy0 - 1 + py, Gx = ox0 - 1 + px;
        const bool ok = (Gy >= 0) & (Gy < Hh) & (Gx >= 0) & (Gx < Hh);
        unsigned short* ap = (unsigned short*)(sm + OFF_U) + p * (ASTR / 2);
        const size_t base = (size_t)b * Cc * HW + (size_t)Gy * 56 + Gx;
        for (int c = 0; c < Cc; ++c) {
            float xv = ok ? x[base + (size_t)c * HW] : 0.f;
            ap[c] = f2bf(xv);
        }
        unsigned short* w1p = (unsigned short*)(sm + OFF_W1);
        for (int i = tid; i < HID * Cc; i += 256) {
            int r = i >> 6, cc = i & 63;
            w1p[r * (W1STR / 2) + cc] = f2bf(w1[i]);
        }
    }
    __syncthreads();

    // ---------------- Stage A: expand GEMM  M=256 px, N=384, K=64 ----------------
    // wave w handles N-tiles 3w..3w+2, all 16 M-tiles. B frags hoisted per nt.
    for (int nt = wv * 3; nt < wv * 3 + 3; ++nt) {
        const int ch = nt * 16 + lrow;
        const float s1  = g1[ch] * rsqrtf(v1[ch] + EPSV);
        const float bb1 = b1[ch] - m1[ch] * s1;
        const char* brow = sm + OFF_W1 + ch * W1STR;
        const v16bf fb0 = lds_frag2(brow + kh * 16,      brow + kh * 16 + 32);
        const v16bf fb1 = lds_frag2(brow + 64 + kh * 16, brow + 96 + kh * 16);
        for (int mt = 0; mt < 16; ++mt) {
            const char* arow = sm + OFF_U + (mt * 16 + lrow) * ASTR;
            const v16bf fa0 = lds_frag2(arow + kh * 16,      arow + kh * 16 + 32);
            const v16bf fa1 = lds_frag2(arow + 64 + kh * 16, arow + 96 + kh * 16);
            v8f acc = {};
            acc = wmma_bf16(fa0, fb0, acc);
            acc = wmma_bf16(fa1, fb1, acc);
#pragma unroll
            for (int r = 0; r < 8; ++r) {
                const int prow = mt * 16 + r + kh * 8;
                float v = fminf(fmaxf(acc[r] * s1 + bb1, 0.f), 6.f);
                hs[prow * 392 + ch] = f2bf(v);
            }
        }
    }
    __syncthreads();

    // ---- stash depthwise weights + precomputed BN2 scale/bias in dead union LDS --
    float* wdl  = (float*)(sm + OFF_U);          // 3456 floats
    float* s2l  = wdl + HID * 9;                 // 384 floats
    float* bb2l = s2l + HID;                     // 384 floats
    for (int i = tid; i < HID * 9; i += 256) wdl[i] = wd[i];
    for (int i = tid; i < HID; i += 256) {
        const float s2 = g2[i] * rsqrtf(v2[i] + EPSV);
        s2l[i]  = s2;
        bb2l[i] = b2[i] - m2[i] * s2;
    }
    __syncthreads();

    // ---------------- Stage B: depthwise 3x3 + BN2 + ReLU6 (in-place in h) -------
    for (int cb = 0; cb < 12; ++cb) {
        float vals[25];
        int n = 0;
        for (int i = tid; i < 196 * 32; i += 256) {
            const int op = i >> 5;
            const int c  = (cb << 5) + (i & 31);
            const int oy = op / 14, ox = op - oy * 14;
            float s = 0.f;
#pragma unroll
            for (int dy = 0; dy < 3; ++dy)
#pragma unroll
                for (int dx = 0; dx < 3; ++dx) {
                    const int Gy = oy0 + oy + dy - 1, Gx = ox0 + ox + dx - 1;
                    const bool ok = (Gy >= 0) & (Gy < Hh) & (Gx >= 0) & (Gx < Hh);
                    float hv = ok ? bf2f(hs[((oy + dy) * 16 + ox + dx) * 392 + c]) : 0.f;
                    s += hv * wdl[c * 9 + dy * 3 + dx];
                }
            vals[n++] = fminf(fmaxf(s * s2l[c] + bb2l[c], 0.f), 6.f);
        }
        __syncthreads();                    // all reads done before in-place writes
        n = 0;
        for (int i = tid; i < 196 * 32; i += 256) {
            const int op = i >> 5;
            const int c  = (cb << 5) + (i & 31);
            const int oy = op / 14, ox = op - oy * 14;
            hs[((oy + 1) * 16 + ox + 1) * 392 + c] = f2bf(vals[n++]);
        }
    }
    __syncthreads();

    // -------- async-copy per-sample kT into LDS union region (ASYNCcnt path) -----
    {
        const unsigned long long base =
            (unsigned long long)(const void*)(kt + (size_t)b * KT_PER_B_US);
        for (unsigned off = tid * 16; off < (unsigned)KT_PER_B_BY; off += 256 * 16) {
            async_g2l_b128(OFF_U + off, off, base);   // global_load_async_to_lds_b128
        }
        wait_asynccnt0();                             // s_wait_asynccnt 0
    }
    __syncthreads();

    // ---------------- Stage C: CondConv GEMM  M=196(->208), N=64, K=384 ----------
    for (int pair = wv; pair < 52; pair += 8) {
        const int mt = pair >> 2, nt = pair & 3;
        const int o  = nt * 16 + lrow;
        const int m  = mt * 16 + lrow;
        const int pix = (m < 196) ? ((1 + m / 14) * 16 + 1 + m % 14) : 0;
        const char* arow = sm + OFF_H + pix * HSTR;
        const char* brow = sm + OFF_U + o * KTSTR;
        v8f acc = {};
#pragma unroll 4
        for (int ks = 0; ks < 12; ++ks) {
            const v16bf fa = lds_frag2(arow + ks * 64 + kh * 16,
                                       arow + ks * 64 + kh * 16 + 32);
            const v16bf fb = lds_frag2(brow + ks * 64 + kh * 16,
                                       brow + ks * 64 + kh * 16 + 32);
            acc = wmma_bf16(fa, fb, acc);
        }
        const float s3  = g3[o] * rsqrtf(v3[o] + EPSV);
        const float bb3 = b3[o] - m3[o] * s3;
        __builtin_prefetch(&x[(size_t)(b * OUP + o) * HW + (size_t)oy0 * 56 + ox0], 0, 0);
#pragma unroll
        for (int r = 0; r < 8; ++r) {
            const int M = mt * 16 + r + kh * 8;
            if (M < 196) {
                const int oy = M / 14, ox = M - oy * 14;
                const size_t idx = (size_t)(b * OUP + o) * HW
                                 + (size_t)(oy0 + oy) * 56 + (ox0 + ox);
                out[idx] = acc[r] * s3 + bb3 + x[idx];   // BN3 + residual
            }
        }
    }
}

// =====================================================================
extern "C" void kernel_launch(void* const* d_in, const int* in_sizes, int n_in,
                              void* d_out, int out_size, void* d_ws, size_t ws_size,
                              hipStream_t stream)
{
    const float* x  = (const float*)d_in[0];
    const float* w1 = (const float*)d_in[1];
    const float* g1 = (const float*)d_in[2];
    const float* b1 = (const float*)d_in[3];
    const float* m1 = (const float*)d_in[4];
    const float* v1 = (const float*)d_in[5];
    const float* wd = (const float*)d_in[6];
    const float* g2 = (const float*)d_in[7];
    const float* b2 = (const float*)d_in[8];
    const float* m2 = (const float*)d_in[9];
    const float* v2 = (const float*)d_in[10];
    const float* we = (const float*)d_in[11];
    const float* g3 = (const float*)d_in[12];
    const float* b3 = (const float*)d_in[13];
    const float* m3 = (const float*)d_in[14];
    const float* v3 = (const float*)d_in[15];
    const float* wr = (const float*)d_in[16];
    const float* br = (const float*)d_in[17];

    unsigned short* kt = (unsigned short*)d_ws;   // [B][OUP][392] bf16 padded, ~3.2 MB

    router_kernel<<<Bn, 256, 0, stream>>>(x, wr, br, we, kt);
    fused_kernel<<<Bn * 16, 256, SMEM_BYTES, stream>>>(
        x, w1, g1, b1, m1, v1, wd, g2, b2, m2, v2,
        kt, g3, b3, m3, v3, (float*)d_out);
}
